// LocalPoolPointnetPPFusion_4dims_67654324846837
// MI455X (gfx1250) — matrix operand order repaired and checked by reading, not compiled, not run on hardware
//
// LocalPoolPointnet encoder for MI455X (gfx1250), bf16 WMMA pipeline.
//
// Design (reasoned from MI455X specs, compile-only):
//  * GEMM chain (~0.5 TFLOP) over M=262144 rows is memory-bound at 23.3 TB/s
//    once matrix math runs on v_wmma_f32_16x16x32_bf16 -> keep activations
//    bf16 (half the HBM bytes), fp32 accumulate, fuse bias/ReLU/residual in
//    the GEMM epilogue so every activation tensor is streamed exactly once.
//  * 128x64 output tile per 256-thread block (8 wave32s, 4 WMMA tiles each).
//    A staged row-major in LDS (lane fragment = 2x ds_load_b128); B staged
//    pair-transposed (BsT[n][k/2] packed bf16 pairs) so B fragments are also
//    2x ds_load_b128.
//  * Double-buffered LDS: stage slab k+1 while WMMAs consume slab k; one
//    barrier per K-slab. global_prefetch_b8 warms slab k+2.
//  * Packed ReLU on bf16 == signed-i16 max vs 0 (sign-magnitude monotonicity)
//    -> v_pk_max_i16, one VALU op per dword (bf16-float max scalarizes).
//  * Pooling = fp32 grids (50 MB, fits 192MB L2) + ordered-int atomic max;
//    scatter mean = fp32 atomics. Tiny state-MLP/plane-index kernels = VALU.
//
// Input pointer order assumes the harness flattens setup_inputs() with dict
// insertion order for the top-level entries (p, state_start, state_end,
// params) and JAX pytree (alphabetical-key) order inside `params`.

#include <hip/hip_runtime.h>
#include <cstdint>
#include <cstddef>

typedef __attribute__((ext_vector_type(16))) __bf16 v16bf;
typedef __attribute__((ext_vector_type(8)))  short  v8s;
typedef __attribute__((ext_vector_type(8)))  float  v8f;

#define NB    8
#define NT    32768
#define NM    (NB * NT)      // 262144 rows
#define RESO  64
#define R2    4096
#define HID   128
#define FEAT  256
#define CDIM  64

#define BM    128            // block tile M
#define BN    64             // block tile N
#define LDA_S 40             // A LDS row stride (bf16): 32 + 8 pad, 80B rows
#define BT_LD 20             // B LDS row stride (packed uint pairs): 16 + 4 pad

// ---------------------------------------------------------------------------
// small utility kernels
// ---------------------------------------------------------------------------
__global__ void k_fill_f32(float* __restrict__ p, float v, int n) {
  int i = blockIdx.x * blockDim.x + threadIdx.x;
  if (i < n) p[i] = v;
}

__global__ void k_cvt_bf16(const float* __restrict__ s, __bf16* __restrict__ d, int n) {
  int i = blockIdx.x * blockDim.x + threadIdx.x;
  if (i < n) d[i] = (__bf16)s[i];
}

// plane indices: xz=(0,2), xy=(0,1), yz=(1,2); ONet normalize+coord2index
__global__ void k_plane_index(const float* __restrict__ p,
                              int* __restrict__ ixz, int* __restrict__ ixy,
                              int* __restrict__ iyz, int n) {
  int t = blockIdx.x * blockDim.x + threadIdx.x;
  if (t >= n) return;
  float x = p[t * 4 + 0], y = p[t * 4 + 1], z = p[t * 4 + 2];
  const float inv = 1.0f / 1.001f;            // 1/(1+PAD+1e-3)
  auto cell = [&](float a, float b) {
    float u = fminf(fmaxf(a * inv + 0.5f, 0.0f), 1.0f - 1e-5f);
    float v = fminf(fmaxf(b * inv + 0.5f, 0.0f), 1.0f - 1e-5f);
    return (int)floorf(u * (float)RESO) + RESO * (int)floorf(v * (float)RESO);
  };
  ixz[t] = cell(x, z);
  ixy[t] = cell(x, y);
  iyz[t] = cell(y, z);
}

// state MLP: scalar -> 64 -> 128 -> 256 (one block per batch element)
__global__ void k_state_mlp(const float* __restrict__ s,
                            const float* __restrict__ W1, const float* __restrict__ b1,
                            const float* __restrict__ W2, const float* __restrict__ b2,
                            const float* __restrict__ W3, const float* __restrict__ b3,
                            float* __restrict__ out, int out_stride, int col_off) {
  __shared__ float h1[64];
  __shared__ float h2[128];
  int b = blockIdx.x;
  int j = threadIdx.x;
  float sv = s[b];
  if (j < 64) h1[j] = fmaxf(sv * W1[j] + b1[j], 0.0f);
  __syncthreads();
  if (j < 128) {
    float acc = b2[j];
    for (int k = 0; k < 64; ++k) acc += h1[k] * W2[k * 128 + j];
    h2[j] = fmaxf(acc, 0.0f);
  }
  __syncthreads();
  float acc = b3[j];
  for (int k = 0; k < 128; ++k) acc += h2[k] * W3[k * 256 + j];
  out[(size_t)b * out_stride + col_off + j] = acc;
}

// feat_pos first layer: h = relu(p @ W1 + b1), K=4 (VALU is fine), bf16 out
__global__ void k_featpos_h(const float* __restrict__ p, const float* __restrict__ W1,
                            const float* __restrict__ b1, __bf16* __restrict__ h) {
  int t = blockIdx.x;
  int c = threadIdx.x;
  float p0 = p[t * 4 + 0], p1 = p[t * 4 + 1], p2 = p[t * 4 + 2], p3 = p[t * 4 + 3];
  float acc = b1[c] + p0 * W1[c] + p1 * W1[256 + c] + p2 * W1[512 + c] + p3 * W1[768 + c];
  h[(size_t)t * 256 + c] = (__bf16)fmaxf(acc, 0.0f);
}

// ---------------------------------------------------------------------------
// WMMA GEMM: out = act(A) @ W + bias (+ addsrc); bf16 A/W, fp32 accumulate
// ---------------------------------------------------------------------------
// Packed bf16 ReLU via signed-i16 max vs 0: sign-magnitude bf16 bit patterns
// are monotone vs int16 on [0,+max] and negative iff int16-negative, so
// smax(bits,0) == relu. Lowers to v_pk_max_i16 (one op per dword).
__device__ __forceinline__ uint4 relu_bf16x8(uint4 x) {
  union { uint4 u; v8s v; } c;
  c.u = x;
  v8s z = {};
  c.v = __builtin_elementwise_max(c.v, z);
  return c.u;
}

// A fragment (16x32 bf16, ISA 7.12.2): lane m = mbase+(lane&15);
// lanes 0-15 hold K {kb=0: 0..7, 16..23}, lanes 16-31 kb=8. Each half is a
// contiguous 16B run in the row-major LDS tile -> two ds_load_b128.
__device__ __forceinline__ v16bf frag_A(const __bf16* As, int mbase) {
  int lane = threadIdx.x & 31;
  int m = mbase + (lane & 15);
  int kb = (lane & 16) ? 8 : 0;
  union { uint4 u[2]; v16bf v; } r;
  r.u[0] = *(const uint4*)&As[m * LDA_S + kb];
  r.u[1] = *(const uint4*)&As[m * LDA_S + 16 + kb];
  return r.v;
}

// B fragment (32x16 bf16): lane n = nbase+(lane&15); lanes 0-15 hold K {0..7, 16..23},
// lanes 16-31 K {8..15, 24..31} as (k,k+1) pairs per VGPR. BsT stores the tile
// pair-transposed: BsT[n][k/2] = bf16pair(k,n | k+1,n), so each lane's
// fragment is two contiguous ds_load_b128.
__device__ __forceinline__ v16bf frag_B(const unsigned* BsT, int nbase) {
  int lane = threadIdx.x & 31;
  int n = nbase + (lane & 15);
  int kp = (lane & 16) ? 4 : 0;
  union { uint4 u[2]; v16bf v; } r;
  r.u[0] = *(const uint4*)&BsT[n * BT_LD + kp];
  r.u[1] = *(const uint4*)&BsT[n * BT_LD + kp + 8];
  return r.v;
}

__global__ void __launch_bounds__(256)
k_gemm_bf16(const __bf16* __restrict__ A, int lda, int relu_in, int relu_out,
            const __bf16* __restrict__ W, int ldw,
            const float* __restrict__ bias,
            const float* __restrict__ addsrc, int ldadd,
            __bf16* __restrict__ outb, int ldob,
            float* __restrict__ outf, int ldof,
            int M, int N, int K) {
  __shared__ __align__(16) __bf16   As[2][BM * LDA_S];   // 2 x 10240 B
  __shared__ __align__(16) unsigned BsT[2][BN * BT_LD];  // 2 x  5120 B
  (void)M; (void)N;

  int tid  = threadIdx.x;
  int lane = tid & 31;
  int wave = tid >> 5;
  int row0 = blockIdx.x * BM;
  int col0 = blockIdx.y * BN;

  int mi0 = (wave & 3) * 2;    // m-subtiles {mi0, mi0+1} of 8
  int ni0 = (wave >> 2) * 2;   // n-subtiles {ni0, ni0+1} of 4

  v8f acc[2][2] = {{{}, {}}, {{}, {}}};
  if (bias) {
    int nl = lane & 15;
    float bv0 = bias[col0 + (ni0 + 0) * 16 + nl];
    float bv1 = bias[col0 + (ni0 + 1) * 16 + nl];
#pragma unroll
    for (int r = 0; r < 8; ++r) {
      acc[0][0][r] = bv0; acc[1][0][r] = bv0;
      acc[0][1][r] = bv1; acc[1][1][r] = bv1;
    }
  }

  // A loader: 128x32 tile, thread -> row tid>>1, 16 elems (two uint4)
  int a_row = tid >> 1, a_chk = (tid & 1) * 16;
  // B loader: 32x64 tile as 16 pair-rows x 64; thread -> kp=tid>>4, n0=(tid&15)*4
  int b_kp = tid >> 4, b_n0 = (tid & 15) * 4;

  // stage one 32-deep K-slab into LDS buffer `buf`
  auto stage = [&](int buf, int kk) {
    const __bf16* ap = A + (size_t)(row0 + a_row) * lda + kk + a_chk;
    uint4 av0 = *(const uint4*)ap;
    uint4 av1 = *(const uint4*)(ap + 8);
    if (relu_in) { av0 = relu_bf16x8(av0); av1 = relu_bf16x8(av1); }
    *reinterpret_cast<uint4*>(&As[buf][a_row * LDA_S + a_chk]) = av0;
    *reinterpret_cast<uint4*>(&As[buf][a_row * LDA_S + a_chk + 8]) = av1;

    const __bf16* wp = W + (size_t)(kk + 2 * b_kp) * ldw + col0 + b_n0;
    uint2 w0 = *(const uint2*)wp;          // row 2kp,   n0..n0+3
    uint2 w1 = *(const uint2*)(wp + ldw);  // row 2kp+1, n0..n0+3
    BsT[buf][(b_n0 + 0) * BT_LD + b_kp] = (w0.x & 0xFFFFu) | (w1.x << 16);
    BsT[buf][(b_n0 + 1) * BT_LD + b_kp] = (w0.x >> 16) | (w1.x & 0xFFFF0000u);
    BsT[buf][(b_n0 + 2) * BT_LD + b_kp] = (w0.y & 0xFFFFu) | (w1.y << 16);
    BsT[buf][(b_n0 + 3) * BT_LD + b_kp] = (w0.y >> 16) | (w1.y & 0xFFFF0000u);
  };

  stage(0, 0);
  for (int k0 = 0; k0 < K; k0 += 32) {
    int cur = (k0 >> 5) & 1;
    __syncthreads();                         // stage(cur) now visible
    if (k0 + 32 < K) stage(cur ^ 1, k0 + 32);  // overlap next slab with WMMAs
    if (k0 + 64 < K) {                       // warm caches 2 slabs ahead
      __builtin_prefetch((const void*)(A + (size_t)(row0 + a_row) * lda + k0 + 64 + a_chk), 0, 1);
      __builtin_prefetch((const void*)(W + (size_t)(k0 + 64 + 2 * b_kp) * ldw + col0 + b_n0), 0, 1);
    }
    v16bf a0 = frag_A(As[cur], (mi0 + 0) * 16);
    v16bf a1 = frag_A(As[cur], (mi0 + 1) * 16);
    v16bf b0 = frag_B(BsT[cur], (ni0 + 0) * 16);
    v16bf b1 = frag_B(BsT[cur], (ni0 + 1) * 16);
    acc[0][0] = __builtin_amdgcn_wmma_f32_16x16x32_bf16(false, a0, false, b0, (short)0, acc[0][0], false, false);
    acc[0][1] = __builtin_amdgcn_wmma_f32_16x16x32_bf16(false, a0, false, b1, (short)0, acc[0][1], false, false);
    acc[1][0] = __builtin_amdgcn_wmma_f32_16x16x32_bf16(false, a1, false, b0, (short)0, acc[1][0], false, false);
    acc[1][1] = __builtin_amdgcn_wmma_f32_16x16x32_bf16(false, a1, false, b1, (short)0, acc[1][1], false, false);
  }

  // C/D layout: VGPR r -> M = r + (lane>=16 ? 8 : 0), N = lane&15
  int nl = lane & 15;
  int mh = (lane & 16) ? 8 : 0;
#pragma unroll
  for (int im = 0; im < 2; ++im) {
#pragma unroll
    for (int jn = 0; jn < 2; ++jn) {
#pragma unroll
      for (int r = 0; r < 8; ++r) {
        size_t m = (size_t)(row0 + (mi0 + im) * 16 + r + mh);
        int    n = col0 + (ni0 + jn) * 16 + nl;
        float  v = acc[im][jn][r];
        if (addsrc)   v += addsrc[m * ldadd + n];
        if (relu_out) v = fmaxf(v, 0.0f);
        if (outf)     outf[m * ldof + n] = v;
        if (outb)     outb[m * ldob + n] = (__bf16)v;
      }
    }
  }
}

// ---------------------------------------------------------------------------
// pooling: per-plane segment max into fp32 grids, then 3-plane gather-sum
// ---------------------------------------------------------------------------
__device__ __forceinline__ void atomicMaxF(float* addr, float v) {
  if (v >= 0.0f) atomicMax((int*)addr, __float_as_int(v));
  else           atomicMin((unsigned int*)addr, __float_as_uint(v));
}

__global__ void k_pool_scatter(const __bf16* __restrict__ x, int ldx,
                               const int* __restrict__ i0, const int* __restrict__ i1,
                               const int* __restrict__ i2, float* __restrict__ grid) {
  int t = blockIdx.x;           // point row
  int f = threadIdx.x;          // feature 0..127
  int b = t >> 15;              // t / NT
  float v = (float)x[(size_t)t * ldx + f];
  size_t base = (size_t)b * R2 * HID;
  const size_t ps = (size_t)NB * R2 * HID;
  atomicMaxF(&grid[0 * ps + base + (size_t)i0[t] * HID + f], v);
  atomicMaxF(&grid[1 * ps + base + (size_t)i1[t] * HID + f], v);
  atomicMaxF(&grid[2 * ps + base + (size_t)i2[t] * HID + f], v);
}

__global__ void k_pool_gather(const float* __restrict__ grid,
                              const int* __restrict__ i0, const int* __restrict__ i1,
                              const int* __restrict__ i2,
                              __bf16* __restrict__ xdst, int ldx) {
  int t = blockIdx.x;
  int f = threadIdx.x;
  int b = t >> 15;
  size_t base = (size_t)b * R2 * HID;
  const size_t ps = (size_t)NB * R2 * HID;
  float v = grid[0 * ps + base + (size_t)i0[t] * HID + f]
          + grid[1 * ps + base + (size_t)i1[t] * HID + f]
          + grid[2 * ps + base + (size_t)i2[t] * HID + f];
  xdst[(size_t)t * ldx + HID + f] = (__bf16)v;   // concat([x, pooled]) cols 128..255
}

// ---------------------------------------------------------------------------
// scatter-mean to output feature planes
// ---------------------------------------------------------------------------
__global__ void k_smean_accum(const float* __restrict__ c, const int* __restrict__ idx,
                              float* __restrict__ sums, float* __restrict__ cnt) {
  int t = blockIdx.x;
  int f = threadIdx.x;          // 0..63
  int b = t >> 15;
  int cell = idx[t];
  atomicAdd(&sums[((size_t)b * R2 + cell) * CDIM + f], c[(size_t)t * CDIM + f]);
  if (f == 0) atomicAdd(&cnt[(size_t)b * R2 + cell], 1.0f);
}

__global__ void k_smean_write(const float* __restrict__ sums, const float* __restrict__ cnt,
                              float* __restrict__ out) {   // out at [plane] base: (B,C,R2)
  int g = blockIdx.x * blockDim.x + threadIdx.x;
  if (g >= NB * R2 * CDIM) return;
  int f = g & (CDIM - 1);
  int cell = (g >> 6) & (R2 - 1);
  int b = g >> 18;
  float n = cnt[(size_t)b * R2 + cell];
  float v = sums[((size_t)b * R2 + cell) * CDIM + f] / fmaxf(n, 1.0f);
  out[((size_t)b * CDIM + f) * R2 + cell] = v;
}

// ---------------------------------------------------------------------------
// host orchestration
// ---------------------------------------------------------------------------
enum {
  IN_P = 0, IN_SSTART, IN_SEND,
  IP_BLKC_W0, IP_BLKC_W1, IP_BLKC_WS, IP_BLKC_B0, IP_BLKC_B1,
  IP_BLKG_W0, IP_BLKG_W1, IP_BLKG_WS, IP_BLKG_B0, IP_BLKG_B1,
  IP_FCCW, IP_FCCB, IP_FCCCW, IP_FCCCB,
  IP_FPW1, IP_FPWC, IP_FPWG, IP_FPB1, IP_FPBC, IP_FPBG,
  IP_L1W, IP_L1B, IP_L2W, IP_L2B, IP_L3W, IP_L3B,
  IP_L4W, IP_L4B, IP_L5W, IP_L5B, IP_L6W, IP_L6B
};

extern "C" void kernel_launch(void* const* d_in, const int* in_sizes, int n_in,
                              void* d_out, int out_size, void* d_ws, size_t ws_size,
                              hipStream_t stream) {
  (void)in_sizes; (void)n_in; (void)out_size; (void)ws_size;

  const float* p      = (const float*)d_in[IN_P];
  const float* sstart = (const float*)d_in[IN_SSTART];
  const float* send   = (const float*)d_in[IN_SEND];
  auto F = [&](int i) { return (const float*)d_in[i]; };

  // ---- workspace layout ----
  char* ws = (char*)d_ws;
  size_t off = 0;
  auto take = [&](size_t bytes) -> char* {
    char* q = ws + off;
    off += (bytes + 255) & ~(size_t)255;
    return q;
  };
  int*    ixz  = (int*)take((size_t)NM * 4);
  int*    ixy  = (int*)take((size_t)NM * 4);
  int*    iyz  = (int*)take((size_t)NM * 4);
  __bf16* wWg  = (__bf16*)take((size_t)65536 * 2);
  __bf16* wWc  = (__bf16*)take((size_t)65536 * 2);
  __bf16* wgW0 = (__bf16*)take((size_t)5 * 256 * 128 * 2);
  __bf16* wgW1 = (__bf16*)take((size_t)5 * 128 * 128 * 2);
  __bf16* wgWs = (__bf16*)take((size_t)5 * 256 * 128 * 2);
  __bf16* wcW0 = (__bf16*)take((size_t)5 * 256 * 128 * 2);
  __bf16* wcW1 = (__bf16*)take((size_t)5 * 128 * 128 * 2);
  __bf16* wcWs = (__bf16*)take((size_t)5 * 256 * 128 * 2);
  __bf16* wFcg = (__bf16*)take((size_t)128 * 64 * 2);
  __bf16* wFcc = (__bf16*)take((size_t)128 * 64 * 2);
  __bf16* X    = (__bf16*)take((size_t)NM * 256 * 2);  // also hosts feat_pos h
  __bf16* Y    = (__bf16*)take((size_t)NM * 256 * 2);
  __bf16* netg = (__bf16*)take((size_t)NM * 256 * 2);
  __bf16* netc = (__bf16*)take((size_t)NM * 256 * 2);
  __bf16* hb   = (__bf16*)take((size_t)NM * 128 * 2);
  float*  tmp  = (float*)take((size_t)NM * 128 * 4);
  float*  grid = (float*)take((size_t)3 * NB * R2 * HID * 4);
  float*  cbuf = (float*)take((size_t)NM * CDIM * 4);
  float*  sums = (float*)take((size_t)NB * R2 * CDIM * 4);
  float*  cnt  = (float*)take((size_t)NB * R2 * 4);

  const int* idx3[3] = {ixz, ixy, iyz};
  float* out = (float*)d_out;
  const size_t PLANE = (size_t)NB * CDIM * R2;   // 2,097,152
  const size_t STATE_OFF = 2 * 3 * PLANE;

  auto cvt = [&](const float* s, __bf16* d, int n) {
    k_cvt_bf16<<<(n + 255) / 256, 256, 0, stream>>>(s, d, n);
  };
  auto fill = [&](float* q, float v, int n) {
    k_fill_f32<<<(n + 255) / 256, 256, 0, stream>>>(q, v, n);
  };
  auto gemm = [&](const __bf16* A, int lda, int relu_in, int relu_out,
                  const __bf16* W, int N, int K,
                  const float* bias, const float* addsrc, int ldadd,
                  __bf16* outb, int ldob, float* outf, int ldof) {
    dim3 g(NM / BM, N / BN);
    k_gemm_bf16<<<g, 256, 0, stream>>>(A, lda, relu_in, relu_out, W, N, bias,
                                       addsrc, ldadd, outb, ldob, outf, ldof,
                                       NM, N, K);
  };

  // ---- weights -> bf16 ----
  cvt(F(IP_FPWG), wWg, 65536);            cvt(F(IP_FPWC), wWc, 65536);
  cvt(F(IP_BLKG_W0), wgW0, 5 * 256 * 128); cvt(F(IP_BLKG_W1), wgW1, 5 * 128 * 128);
  cvt(F(IP_BLKG_WS), wgWs, 5 * 256 * 128);
  cvt(F(IP_BLKC_W0), wcW0, 5 * 256 * 128); cvt(F(IP_BLKC_W1), wcW1, 5 * 128 * 128);
  cvt(F(IP_BLKC_WS), wcWs, 5 * 256 * 128);
  cvt(F(IP_FCCW), wFcg, 128 * 64);        cvt(F(IP_FCCCW), wFcc, 128 * 64);

  // ---- state MLP -> state_feat (B, 512) ----
  k_state_mlp<<<NB, 256, 0, stream>>>(sstart, F(IP_L1W), F(IP_L1B), F(IP_L2W), F(IP_L2B),
                                      F(IP_L3W), F(IP_L3B), out + STATE_OFF, 512, 0);
  k_state_mlp<<<NB, 256, 0, stream>>>(send, F(IP_L4W), F(IP_L4B), F(IP_L5W), F(IP_L5B),
                                      F(IP_L6W), F(IP_L6B), out + STATE_OFF, 512, 256);

  // ---- plane indices ----
  k_plane_index<<<NM / 256, 256, 0, stream>>>(p, ixz, ixy, iyz, NM);

  // ---- feat_pos: h = relu(p@W1+b1); net/net_corr = h@{Wg,Wc}+b ----
  k_featpos_h<<<NM, 256, 0, stream>>>(p, F(IP_FPW1), F(IP_FPB1), X);   // X == h
  gemm(X, 256, 0, 0, wWg, 256, 256, F(IP_FPBG), nullptr, 0, netg, 256, nullptr, 0);
  gemm(X, 256, 0, 0, wWc, 256, 256, F(IP_FPBC), nullptr, 0, netc, 256, nullptr, 0);

  // ---- one branch: 5 resnet blocks + pooling + final fc + scatter-mean ----
  auto branch = [&](const __bf16* net, const __bf16* bW0, const __bf16* bW1,
                    const __bf16* bWs, const float* bb0, const float* bb1,
                    const __bf16* fcW, const float* fcb, float* outbase) {
    const __bf16* cur = net;
    __bf16* nxt = X;
    for (int i = 0; i < 5; ++i) {
      const __bf16* W0 = bW0 + (size_t)i * 256 * 128;
      const __bf16* W1 = bW1 + (size_t)i * 128 * 128;
      const __bf16* Wsh = bWs + (size_t)i * 256 * 128;
      const float* b0 = bb0 + i * 128;
      const float* b1 = bb1 + i * 128;
      if (i > 0) {  // x2 = concat(x, pool_local(x)) written into cur[:,128:256]
        fill(grid, -__builtin_huge_valf(), 3 * NB * R2 * HID);
        k_pool_scatter<<<NM, 128, 0, stream>>>(cur, 256, ixz, ixy, iyz, grid);
        k_pool_gather<<<NM, 128, 0, stream>>>(grid, ixz, ixy, iyz,
                                              (__bf16*)cur, 256);
      }
      // hb = relu(fc0(relu(x2))) (ReLU folded into epilogue);
      // dx = fc1(hb); y = x2@Ws + dx
      gemm(cur, 256, 1, 1, W0, 128, 256, b0, nullptr, 0, hb, 128, nullptr, 0);
      gemm(hb, 128, 0, 0, W1, 128, 128, b1, nullptr, 0, nullptr, 0, tmp, 128);
      gemm(cur, 256, 0, 0, Wsh, 128, 256, nullptr, tmp, 128, nxt, 256, nullptr, 0);
      cur = nxt;
      nxt = (cur == X) ? Y : X;
    }
    // c = x @ fcW + fcb  (M,64) fp32
    gemm(cur, 256, 0, 0, fcW, 64, 128, fcb, nullptr, 0, nullptr, 0, cbuf, 64);
    // scatter-mean per plane -> (plane, B, C, R, R)
    for (int pl = 0; pl < 3; ++pl) {
      fill(sums, 0.0f, NB * R2 * CDIM);
      fill(cnt, 0.0f, NB * R2);
      k_smean_accum<<<NM, 64, 0, stream>>>(cbuf, idx3[pl], sums, cnt);
      k_smean_write<<<(NB * R2 * CDIM + 255) / 256, 256, 0, stream>>>(
          sums, cnt, outbase + (size_t)pl * PLANE);
    }
  };

  branch(netg, wgW0, wgW1, wgWs, F(IP_BLKG_B0), F(IP_BLKG_B1),
         wFcg, F(IP_FCCB), out);                       // fea_geo
  branch(netc, wcW0, wcW1, wcWs, F(IP_BLKC_B0), F(IP_BLKC_B1),
         wFcc, F(IP_FCCCB), out + 3 * PLANE);          // fea_corr
}
